// Sender_31756988187001
// MI455X (gfx1250) — compile-verified, matrix operand order checked by reading, not run on hardware
//
#include <hip/hip_runtime.h>
#include <hip/hip_bf16.h>
#include <math.h>

// ---------------------------------------------------------------------------
// Sender (Gumbel-softmax LSTM) for MI455X / gfx1250.
// GEMMs run through v_wmma_f32_16x16x32_bf16 (bf16 inputs, f32 accum).
// Per-step: gather -> gates GEMM (WMMA) -> LSTM cell -> scores GEMM (WMMA)
//           -> fused softmax/argmax/CE sampler.
// ---------------------------------------------------------------------------

typedef __bf16 bf16;
typedef __attribute__((ext_vector_type(16))) __bf16 v16bf;
typedef __attribute__((ext_vector_type(8)))  float  v8f;

constexpr int B_  = 256;
constexpr int V_  = 8192;
constexpr int E_  = 256;
constexpr int H_  = 512;
constexpr int NI_ = 2048;
constexpr int L_  = 16;
constexpr int BOUND_ = V_ - 1;
constexpr float TAU_ = 1.2f;

union Frag16 { v16bf v; uint4 q[2]; };

// Per-lane 16-bit fragment load (A 16x32 / B 32x16 layouts, wave32):
// lanes 0-15 hold K = {k0..k0+7, k0+16..k0+23}; lanes 16-31 hold the other
// halves.  Caller passes p = rowbase + k0 + (lane>>4)*8.  Two 16B loads.
__device__ __forceinline__ v16bf load_frag(const bf16* p) {
  Frag16 f;
  f.q[0] = *reinterpret_cast<const uint4*>(p);
  f.q[1] = *reinterpret_cast<const uint4*>(p + 16);
  return f.v;
}

__device__ __forceinline__ v8f wmma_bf16(v16bf a, v16bf b, v8f c) {
  return __builtin_amdgcn_wmma_f32_16x16x32_bf16(
      /*neg_a=*/false, a, /*neg_b=*/false, b,
      /*c_mod=*/(short)0, c, /*reuse_a=*/false, /*reuse_b=*/false);
}

// Accumulate a 32x64 tile: C[m0:m0+32, n0:n0+64] += A[m0:,:K] @ B[n0:,:K]^T
// (both A and B are row-major with K contiguous -> X @ W^T pattern).
__device__ __forceinline__ void gemm_accum(
    v8f (&acc)[2][4],
    const bf16* __restrict__ A, const bf16* __restrict__ Bm,
    int K, int m0, int n0, int lane) {
  const int half = lane >> 4;
  const int sub  = lane & 15;
  const bf16* a0p = A  + (size_t)(m0 + sub)      * K + half * 8;
  const bf16* a1p = A  + (size_t)(m0 + 16 + sub) * K + half * 8;
  const bf16* b0p = Bm + (size_t)(n0 + sub)      * K + half * 8;
  for (int k0 = 0; k0 < K; k0 += 32) {
    v16bf a0 = load_frag(a0p + k0);
    v16bf a1 = load_frag(a1p + k0);
#pragma unroll
    for (int t = 0; t < 4; ++t) {
      v16bf bfr = load_frag(b0p + (size_t)t * 16 * K + k0);
      acc[0][t] = wmma_bf16(a0, bfr, acc[0][t]);
      acc[1][t] = wmma_bf16(a1, bfr, acc[1][t]);
    }
  }
}

// C[M,N] = A1@B1^T (+ A2@B2^T) + bias1 + bias2.  Block = 8 waves along N
// (512 cols), each wave owns a 32x64 tile.  grid = (M/32, N/512).
__global__ __launch_bounds__(256) void gemm_kernel(
    float* __restrict__ C,
    const bf16* __restrict__ A1, const bf16* __restrict__ B1, int K1,
    const bf16* __restrict__ A2, const bf16* __restrict__ B2, int K2,
    const float* __restrict__ bias1, const float* __restrict__ bias2, int N) {
  const int lane = threadIdx.x & 31;
  const int wave = threadIdx.x >> 5;
  const int m0 = blockIdx.x * 32;
  const int n0 = blockIdx.y * 512 + wave * 64;
  if (n0 >= N) return;                       // wave-uniform: EXEC stays full
  v8f acc[2][4] = {};
  gemm_accum(acc, A1, B1, K1, m0, n0, lane);
  if (A2) gemm_accum(acc, A2, B2, K2, m0, n0, lane);
  const int col = lane & 15;
  const int rh  = lane >> 4;
#pragma unroll
  for (int t = 0; t < 4; ++t) {
    int n = n0 + t * 16 + col;
    float bsum = (bias1 ? bias1[n] : 0.f) + (bias2 ? bias2[n] : 0.f);
#pragma unroll
    for (int r = 0; r < 8; ++r) {
      int m = m0 + r + 8 * rh;               // C/D layout: VGPR r, lane half
      C[(size_t)m        * N + n] = acc[0][t][r] + bsum;
      C[(size_t)(m + 16) * N + n] = acc[1][t][r] + bsum;
    }
  }
}

// ---------------------------------------------------------------------------

__global__ void cvt_kernel(bf16* __restrict__ dst, const float* __restrict__ src,
                           size_t n) {
  size_t i = (size_t)blockIdx.x * blockDim.x + threadIdx.x;
  size_t stride = (size_t)gridDim.x * blockDim.x;
  for (; i < n; i += stride) dst[i] = (bf16)src[i];
}

__global__ void fzero_kernel(float* __restrict__ p, size_t n) {
  size_t i = (size_t)blockIdx.x * blockDim.x + threadIdx.x;
  size_t stride = (size_t)gridDim.x * blockDim.x;
  for (; i < n; i += stride) p[i] = 0.f;
}

__global__ void normwc_kernel(const float* __restrict__ wc, float* __restrict__ nwc) {
  __shared__ float red[256];
  __shared__ float total;
  int tid = threadIdx.x;
  float s = 0.f;
  for (int j = tid; j < V_; j += 256) s += wc[j];   // BOUND_WEIGHT == 1.0
  red[tid] = s;
  for (int off = 128; off > 0; off >>= 1) {
    __syncthreads();
    if (tid < off) red[tid] += red[tid + off];
  }
  __syncthreads();
  if (tid == 0) total = red[0];
  __syncthreads();
  for (int j = tid; j < V_; j += 256) nwc[j] = wc[j] / total;
}

__global__ void init_state_kernel(float* __restrict__ msg, int* __restrict__ idx,
                                  int* __restrict__ seq, float* __restrict__ vl,
                                  float* __restrict__ tokv) {
  int b = blockIdx.x * blockDim.x + threadIdx.x;
  if (b < B_) {
    msg[(size_t)b * (L_ + 1) * V_ + BOUND_] = 1.0f;  // tok0 one-hot at BOUND
    idx[b]  = BOUND_;
    seq[b]  = L_ + 1;
    vl[b]   = 0.f;
    tokv[b] = 1.f;
  }
}

// emb_bf[b,:] = bf16(tokval[b] * embedding[idx[b],:])  (one-hot @ embedding)
__global__ __launch_bounds__(256) void gather_kernel(
    const float* __restrict__ embedding, const int* __restrict__ idx,
    const float* __restrict__ tokv, bf16* __restrict__ out) {
  int i = blockIdx.x * 256 + threadIdx.x;
  if (i >= B_ * E_) return;
  int b = i >> 8;           // E_ == 256
  int e = i & 255;
  out[i] = (bf16)(tokv[b] * embedding[(size_t)idx[b] * E_ + e]);
}

__global__ __launch_bounds__(256) void lstm_kernel(
    const float* __restrict__ gates, float* __restrict__ c,
    bf16* __restrict__ h_bf) {
  int i = blockIdx.x * 256 + threadIdx.x;
  if (i >= B_ * H_) return;
  int b = i / H_, j = i - b * H_;
  const float* g = gates + (size_t)b * 4 * H_;
  float gi = g[j], gf = g[H_ + j], gg = g[2 * H_ + j], go = g[3 * H_ + j];
  float si = 1.f / (1.f + expf(-gi));
  float sf = 1.f / (1.f + expf(-gf));
  float so = 1.f / (1.f + expf(-go));
  float cn = sf * c[i] + si * tanhf(gg);
  c[i] = cn;
  h_bf[i] = (bf16)(so * tanhf(cn));
}

// One block per batch row.  Single-pass online logsumexp for
//   u = scores + gumbel   (tau-scaled, with first-occurrence argmax)
//   v = scores - norm_wc  (CE denominator)
// y at argmax == 1/sum_exp_u (shift-invariance kills the first log_softmax).
__global__ __launch_bounds__(256) void sample_kernel(
    const float* __restrict__ scores, const float* __restrict__ gum,
    const float* __restrict__ nwc, int* __restrict__ idx, int* __restrict__ seq,
    float* __restrict__ vl, float* __restrict__ tokv, float* __restrict__ msg,
    int l) {
  __shared__ float mu_s[256], su_s[256], mv_s[256], sv_s[256];
  __shared__ int au_s[256];
  const int b = blockIdx.x, tid = threadIdx.x;
  const float invtau = 1.0f / TAU_;
  const float* s = scores + (size_t)b * V_;
  const float* g = gum + (size_t)b * V_;
  float mu = -INFINITY, su = 0.f, mv = -INFINITY, sv = 0.f;
  int au = V_;
  for (int j = tid; j < V_; j += 256) {
    float sj = s[j];
    float u = sj + g[j];
    if (u > mu) { su = su * expf((mu - u) * invtau) + 1.f; mu = u; au = j; }
    else        { su += expf((u - mu) * invtau); }
    float v = sj - nwc[j];
    if (v > mv) { sv = sv * expf(mv - v) + 1.f; mv = v; }
    else        { sv += expf(v - mv); }
  }
  mu_s[tid] = mu; su_s[tid] = su; au_s[tid] = au; mv_s[tid] = mv; sv_s[tid] = sv;
  for (int off = 128; off > 0; off >>= 1) {
    __syncthreads();
    if (tid < off) {
      float m2 = mu_s[tid + off], s2 = su_s[tid + off]; int a2 = au_s[tid + off];
      float m1 = mu_s[tid],       s1 = su_s[tid];       int a1 = au_s[tid];
      if (m2 > m1)      { mu_s[tid] = m2; su_s[tid] = s2 + s1 * expf((m1 - m2) * invtau); au_s[tid] = a2; }
      else if (m1 > m2) { su_s[tid] = s1 + s2 * expf((m2 - m1) * invtau); }
      else              { su_s[tid] = s1 + s2; au_s[tid] = a1 < a2 ? a1 : a2; }
      float n2 = mv_s[tid + off], t2 = sv_s[tid + off];
      float n1 = mv_s[tid],       t1 = sv_s[tid];
      if (n2 > n1)      { mv_s[tid] = n2; sv_s[tid] = t2 + t1 * expf(n1 - n2); }
      else if (n1 > n2) { sv_s[tid] = t1 + t2 * expf(n2 - n1); }
      else              { sv_s[tid] = t1 + t2; }
    }
  }
  __syncthreads();
  if (tid == 0) {
    int a = au_s[0];
    float ymax = 1.0f / su_s[0];            // exp(0)/sum at the argmax
    float tv = (1.0f - ymax) + ymax;        // straight-through value
    float lse_v = mv_s[0] + logf(sv_s[0]);
    float ce = lse_v - (s[a] - nwc[a]);
    vl[b] += ce;
    idx[b] = a;
    tokv[b] = tv;
    msg[(size_t)b * (L_ + 1) * V_ + (size_t)(l + 1) * V_ + a] = tv;
    if (a == BOUND_ && tv == 1.0f && seq[b] == L_ + 1) seq[b] = l + 2;
  }
}

__global__ void final_kernel(const int* __restrict__ seq,
                             const float* __restrict__ vl,
                             float* __restrict__ out_tail) {
  int b = blockIdx.x * blockDim.x + threadIdx.x;
  if (b < B_) {
    out_tail[b]      = (float)seq[b];
    out_tail[B_ + b] = vl[b];
  }
}

// ---------------------------------------------------------------------------

extern "C" void kernel_launch(void* const* d_in, const int* in_sizes, int n_in,
                              void* d_out, int out_size, void* d_ws, size_t ws_size,
                              hipStream_t stream) {
  (void)in_sizes; (void)n_in; (void)out_size; (void)ws_size;
  const float* t_in      = (const float*)d_in[0];
  const float* wc        = (const float*)d_in[1];
  const float* embedding = (const float*)d_in[2];
  const float* aff_W     = (const float*)d_in[3];
  const float* aff_b     = (const float*)d_in[4];
  const float* W_ih      = (const float*)d_in[5];
  const float* W_hh      = (const float*)d_in[6];
  const float* b_ih      = (const float*)d_in[7];
  const float* b_hh      = (const float*)d_in[8];
  const float* Wp        = (const float*)d_in[9];
  const float* bp        = (const float*)d_in[10];
  const float* gumbel    = (const float*)d_in[11];

  char* w = (char*)d_ws;
  auto alloc = [&](size_t bytes) -> void* {
    void* p = (void*)w;
    w += (bytes + 255) & ~(size_t)255;
    return p;
  };
  bf16*  t_bf    = (bf16*)alloc((size_t)B_ * NI_ * 2);
  bf16*  affW_bf = (bf16*)alloc((size_t)H_ * NI_ * 2);
  bf16*  Wih_bf  = (bf16*)alloc((size_t)4 * H_ * E_ * 2);
  bf16*  Whh_bf  = (bf16*)alloc((size_t)4 * H_ * H_ * 2);
  bf16*  Wp_bf   = (bf16*)alloc((size_t)V_ * H_ * 2);
  bf16*  emb_bf  = (bf16*)alloc((size_t)B_ * E_ * 2);
  bf16*  h_bf    = (bf16*)alloc((size_t)B_ * H_ * 2);
  float* h_f     = (float*)alloc((size_t)B_ * H_ * 4);
  float* c       = (float*)alloc((size_t)B_ * H_ * 4);
  float* gates   = (float*)alloc((size_t)B_ * 4 * H_ * 4);
  float* scores  = (float*)alloc((size_t)B_ * V_ * 4);
  float* nwc     = (float*)alloc((size_t)V_ * 4);
  int*   idx     = (int*)alloc(B_ * 4);
  int*   seq     = (int*)alloc(B_ * 4);
  float* vl      = (float*)alloc(B_ * 4);
  float* tokv    = (float*)alloc(B_ * 4);

  float* msg = (float*)d_out;
  const size_t msgN = (size_t)B_ * (L_ + 1) * V_;

  // --- one-time per call: bf16 weight conversion, norm_wc, init state ---
  cvt_kernel<<<4096, 256, 0, stream>>>(t_bf,    t_in,  (size_t)B_ * NI_);
  cvt_kernel<<<4096, 256, 0, stream>>>(affW_bf, aff_W, (size_t)H_ * NI_);
  cvt_kernel<<<4096, 256, 0, stream>>>(Wih_bf,  W_ih,  (size_t)4 * H_ * E_);
  cvt_kernel<<<4096, 256, 0, stream>>>(Whh_bf,  W_hh,  (size_t)4 * H_ * H_);
  cvt_kernel<<<8192, 256, 0, stream>>>(Wp_bf,   Wp,    (size_t)V_ * H_);
  normwc_kernel<<<1, 256, 0, stream>>>(wc, nwc);
  fzero_kernel<<<4096, 256, 0, stream>>>(msg, msgN);
  fzero_kernel<<<512, 256, 0, stream>>>(c, (size_t)B_ * H_);
  init_state_kernel<<<1, 256, 0, stream>>>(msg, idx, seq, vl, tokv);

  // h0 = t @ aff_W^T + aff_b   (M=256, N=512, K=2048)
  gemm_kernel<<<dim3(B_ / 32, H_ / 512), 256, 0, stream>>>(
      h_f, t_bf, affW_bf, NI_, nullptr, nullptr, 0, aff_b, nullptr, H_);
  cvt_kernel<<<(B_ * H_ + 255) / 256, 256, 0, stream>>>(h_bf, h_f, (size_t)B_ * H_);

  for (int l = 0; l < L_; ++l) {
    gather_kernel<<<(B_ * E_ + 255) / 256, 256, 0, stream>>>(
        embedding, idx, tokv, emb_bf);
    // gates = emb@W_ih^T + h@W_hh^T + b_ih + b_hh  (M=256, N=2048)
    gemm_kernel<<<dim3(B_ / 32, (4 * H_) / 512), 256, 0, stream>>>(
        gates, emb_bf, Wih_bf, E_, h_bf, Whh_bf, H_, b_ih, b_hh, 4 * H_);
    lstm_kernel<<<(B_ * H_ + 255) / 256, 256, 0, stream>>>(gates, c, h_bf);
    // scores = h@Wp^T + bp  (M=256, N=8192, K=512) -- dominant GEMM
    gemm_kernel<<<dim3(B_ / 32, V_ / 512), 256, 0, stream>>>(
        scores, h_bf, Wp_bf, H_, nullptr, nullptr, 0, bp, nullptr, V_);
    sample_kernel<<<B_, 256, 0, stream>>>(
        scores, gumbel + (size_t)l * B_ * V_, nwc, idx, seq, vl, tokv, msg, l);
  }

  final_kernel<<<1, 256, 0, stream>>>(seq, vl, msg + msgN);
}
